// DecoderRNN_76673756168902
// MI455X (gfx1250) — compile-verified, matrix operand order checked by reading, not run on hardware
//
#include <hip/hip_runtime.h>

#define BB 64
#define EE 512
#define HH 1024
#define SS 400
#define VV 50000
#define EXTV 50100

typedef __attribute__((ext_vector_type(16))) __bf16          bfrag16;
typedef __attribute__((ext_vector_type(16))) unsigned short  usfrag16;
typedef __attribute__((ext_vector_type(8)))  unsigned int    v8u;
typedef __attribute__((ext_vector_type(8)))  float           v8f;

__device__ __forceinline__ unsigned short f2bf(float f) {
  unsigned int x = __float_as_uint(f);
  x += 0x7FFFu + ((x >> 16) & 1u);           // round-to-nearest-even
  return (unsigned short)(x >> 16);
}

__device__ __forceinline__ bfrag16 frag_u(uint4 a, uint4 b) {
  v8u u;
  u[0]=a.x; u[1]=a.y; u[2]=a.z; u[3]=a.w;
  u[4]=b.x; u[5]=b.y; u[6]=b.z; u[7]=b.w;
  return __builtin_bit_cast(bfrag16, u);
}

__device__ __forceinline__ bfrag16 frag_f(float4 w0, float4 w1, float4 w2, float4 w3) {
  usfrag16 u;
  u[0]=f2bf(w0.x); u[1]=f2bf(w0.y); u[2]=f2bf(w0.z); u[3]=f2bf(w0.w);
  u[4]=f2bf(w1.x); u[5]=f2bf(w1.y); u[6]=f2bf(w1.z); u[7]=f2bf(w1.w);
  u[8]=f2bf(w2.x); u[9]=f2bf(w2.y); u[10]=f2bf(w2.z); u[11]=f2bf(w2.w);
  u[12]=f2bf(w3.x); u[13]=f2bf(w3.y); u[14]=f2bf(w3.z); u[15]=f2bf(w3.w);
  return __builtin_bit_cast(bfrag16, u);
}

__device__ __forceinline__ float sigf(float x) { return 1.0f / (1.0f + expf(-x)); }

// Per-lane async copy global -> LDS, 16 bytes, tracked with ASYNCcnt (CDNA5).
// lds_off is the LDS byte offset (low 32 bits of a generic shared pointer).
__device__ __forceinline__ void async_g2l_b128(unsigned lds_off, const void* gaddr) {
  asm volatile("global_load_async_to_lds_b128 %0, %1, off"
               :
               : "v"(lds_off), "v"(gaddr)
               : "memory");
}

__device__ __forceinline__ void wait_async0() {
  asm volatile("s_wait_asynccnt 0x0" ::: "memory");
}

// ---------------- fp32 -> bf16 conversion ----------------
__global__ void k_conv(const float* __restrict__ src, unsigned short* __restrict__ dst, int n) {
  int i = blockIdx.x * blockDim.x + threadIdx.x;
  if (i < n) dst[i] = f2bf(src[i]);
}

// ---------------- dec_in = concat(emb, prev_ctx) @ W_fc^T + b_fc ----------------
__global__ void k_decin(const float* __restrict__ emb, const float* __restrict__ pctx,
                        const float* __restrict__ Wfc, const float* __restrict__ bfc,
                        float* __restrict__ dec_in) {
  int i = blockIdx.x * blockDim.x + threadIdx.x;
  if (i >= EE * 8) return;
  int e = i >> 3, grp = i & 7;
  const float* w = Wfc + (size_t)e * (HH + EE);
  float acc[8];
#pragma unroll
  for (int q = 0; q < 8; ++q) acc[q] = bfc[e];
  for (int k = 0; k < EE; ++k) {
    float wv = w[k];
#pragma unroll
    for (int q = 0; q < 8; ++q) acc[q] += emb[(grp * 8 + q) * EE + k] * wv;
  }
  for (int k = 0; k < HH; ++k) {
    float wv = w[EE + k];
#pragma unroll
    for (int q = 0; q < 8; ++q) acc[q] += pctx[(grp * 8 + q) * HH + k] * wv;
  }
#pragma unroll
  for (int q = 0; q < 8; ++q) dec_in[(grp * 8 + q) * EE + e] = acc[q];
}

// ---------------- gates = dec_in @ W_ih^T + h0 @ W_hh^T + b ----------------
__global__ void k_gates(const float* __restrict__ dec_in, const float* __restrict__ h0,
                        const float* __restrict__ Wih, const float* __restrict__ Whh,
                        const float* __restrict__ bih, const float* __restrict__ bhh,
                        float* __restrict__ gates) {
  int i = blockIdx.x * blockDim.x + threadIdx.x;
  if (i >= 4 * HH * 8) return;
  int j = i >> 3, grp = i & 7;
  const float* wi = Wih + (size_t)j * EE;
  const float* wh = Whh + (size_t)j * HH;
  float acc[8];
#pragma unroll
  for (int q = 0; q < 8; ++q) acc[q] = bih[j] + bhh[j];
  for (int k = 0; k < EE; ++k) {
    float wv = wi[k];
#pragma unroll
    for (int q = 0; q < 8; ++q) acc[q] += dec_in[(grp * 8 + q) * EE + k] * wv;
  }
  for (int k = 0; k < HH; ++k) {
    float wv = wh[k];
#pragma unroll
    for (int q = 0; q < 8; ++q) acc[q] += h0[(grp * 8 + q) * HH + k] * wv;
  }
#pragma unroll
  for (int q = 0; q < 8; ++q) gates[(size_t)(grp * 8 + q) * 4 * HH + j] = acc[q];
}

// ---------------- LSTM pointwise; hidden = concat(h, c) ----------------
__global__ void k_lstm(const float* __restrict__ gates, const float* __restrict__ c0,
                       float* __restrict__ hidden, float* __restrict__ outH, float* __restrict__ outC) {
  int i = blockIdx.x * blockDim.x + threadIdx.x;
  if (i >= BB * HH) return;
  int b = i / HH, h = i - b * HH;
  const float* g = gates + (size_t)b * 4 * HH;
  float ig = g[h], fg = g[HH + h], gg = g[2 * HH + h], og = g[3 * HH + h];
  float c = sigf(fg) * c0[i] + sigf(ig) * tanhf(gg);
  float hv = sigf(og) * tanhf(c);
  hidden[(size_t)b * 2 * HH + h]      = hv;
  hidden[(size_t)b * 2 * HH + HH + h] = c;
  outH[i] = hv;
  outC[i] = c;
}

// ---------------- q_proj = hidden @ Wq^T ----------------
__global__ void k_qproj(const float* __restrict__ hidden, const float* __restrict__ Wq,
                        float* __restrict__ qproj) {
  int i = blockIdx.x * blockDim.x + threadIdx.x;
  if (i >= HH * 8) return;
  int e = i >> 3, grp = i & 7;
  const float* w = Wq + (size_t)e * 2 * HH;
  float acc[8];
#pragma unroll
  for (int q = 0; q < 8; ++q) acc[q] = 0.f;
  for (int k = 0; k < 2 * HH; ++k) {
    float wv = w[k];
#pragma unroll
    for (int q = 0; q < 8; ++q) acc[q] += hidden[(size_t)(grp * 8 + q) * 2 * HH + k] * wv;
  }
#pragma unroll
  for (int q = 0; q < 8; ++q) qproj[(size_t)(grp * 8 + q) * HH + e] = acc[q];
}

// ---------------- energy[b,s] = v . tanh(qproj[b] + enc[s,b] @ Wm^T + b_attn + cov*w) ----------------
// One block per s. 64x1024 bf16 tile of encoder_hiddens[s] staged into 128KB LDS
// with GLOBAL_LOAD_ASYNC_TO_LDS_B128 (ASYNCcnt-tracked, no VGPR round trip).
__global__ __launch_bounds__(256) void k_energy(
    const unsigned short* __restrict__ encb, const unsigned short* __restrict__ wmb,
    const float* __restrict__ qproj, const float* __restrict__ battn,
    const float* __restrict__ vattn, const float* __restrict__ coverw,
    const float* __restrict__ coverage, const int* __restrict__ mask,
    float* __restrict__ energy) {
  extern __shared__ char smem[];
  unsigned short* Alds = (unsigned short*)smem;               // 64*1024 bf16 = 128KB
  float* ered = (float*)(smem + (size_t)BB * HH * 2);         // 8 waves * 64 rows
  const int s = blockIdx.x;
  {
    const char* g = (const char*)(encb + (size_t)s * BB * HH);
    // low 32 bits of a generic shared-aperture address = LDS byte offset
    unsigned ldsbase = (unsigned)(unsigned long long)(void*)Alds;
    for (int i = threadIdx.x; i < BB * HH / 8; i += 256)
      async_g2l_b128(ldsbase + (unsigned)i * 16u, g + (size_t)i * 16);
    wait_async0();
  }
  __syncthreads();
  const int wave = threadIdx.x >> 5;
  const int lane = threadIdx.x & 31;
  const int hi = lane >> 4;
  const int n = lane & 15;
  // coverage term depends only on this lane's 32 fixed rows; hoist out of nt loop
  float covv[4][8];
#pragma unroll
  for (int t = 0; t < 4; ++t)
#pragma unroll
    for (int r = 0; r < 8; ++r)
      covv[t][r] = coverage[(size_t)(t * 16 + 8 * hi + r) * SS + s];
  float pacc[4][8];
#pragma unroll
  for (int t = 0; t < 4; ++t)
#pragma unroll
    for (int r = 0; r < 8; ++r) pacc[t][r] = 0.f;

  for (int nt = wave; nt < HH / 16; nt += 8) {
    const int ecol = nt * 16 + n;
    const uint4* wrow = (const uint4*)(wmb + (size_t)ecol * HH);
    v8f acc[4];
#pragma unroll
    for (int t = 0; t < 4; ++t) {
      v8f z = {0.f, 0.f, 0.f, 0.f, 0.f, 0.f, 0.f, 0.f};
      acc[t] = z;
    }
    for (int kc = 0; kc < HH / 32; ++kc) {
      const int kb = kc * 32;
      bfrag16 bfr = frag_u(wrow[(kb + 8 * hi) >> 3], wrow[(kb + 16 + 8 * hi) >> 3]);
#pragma unroll
      for (int t = 0; t < 4; ++t) {
        const uint4* arow = (const uint4*)(Alds + (size_t)(t * 16 + n) * HH);
        bfrag16 afr = frag_u(arow[(kb + 8 * hi) >> 3], arow[(kb + 16 + 8 * hi) >> 3]);
        acc[t] = __builtin_amdgcn_wmma_f32_16x16x32_bf16(
            false, afr, false, bfr, (short)0, acc[t], false, false);
      }
    }
    const float ba = battn[ecol];
    const float cwv = coverw[ecol];
    const float va = vattn[ecol];
#pragma unroll
    for (int t = 0; t < 4; ++t) {
#pragma unroll
      for (int r = 0; r < 8; ++r) {
        const int b = t * 16 + 8 * hi + r;
        float x = acc[t][r] + qproj[(size_t)b * HH + ecol] + ba + covv[t][r] * cwv;
        pacc[t][r] += tanhf(x) * va;
      }
    }
  }
  // reduce over the 16 lanes of the N dimension (stays within each hi-half)
#pragma unroll
  for (int t = 0; t < 4; ++t) {
#pragma unroll
    for (int r = 0; r < 8; ++r) {
      float v = pacc[t][r];
      v += __shfl_xor(v, 1, 32);
      v += __shfl_xor(v, 2, 32);
      v += __shfl_xor(v, 4, 32);
      v += __shfl_xor(v, 8, 32);
      pacc[t][r] = v;
    }
  }
  if (n == 0) {
#pragma unroll
    for (int t = 0; t < 4; ++t)
#pragma unroll
      for (int r = 0; r < 8; ++r)
        ered[wave * BB + t * 16 + 8 * hi + r] = pacc[t][r];
  }
  __syncthreads();
  if (threadIdx.x < BB) {
    float e = 0.f;
#pragma unroll
    for (int w = 0; w < 8; ++w) e += ered[w * BB + threadIdx.x];
    const int b = threadIdx.x;
    if (!(mask[(size_t)b * SS + s] > 0)) e = -1e12f;
    energy[(size_t)b * SS + s] = e;
  }
}

// ---------------- softmax over S -> enc_attn ----------------
__global__ __launch_bounds__(512) void k_attnsm(const float* __restrict__ energy,
                                                float* __restrict__ attn) {
  __shared__ float red[512];
  int b = blockIdx.x, t = threadIdx.x;
  float v = (t < SS) ? energy[(size_t)b * SS + t] : -3.0e38f;
  red[t] = v;
  __syncthreads();
  for (int off = 256; off > 0; off >>= 1) {
    if (t < off) red[t] = fmaxf(red[t], red[t + off]);
    __syncthreads();
  }
  float mx = red[0];
  __syncthreads();
  float ex = (t < SS) ? expf(v - mx) : 0.f;
  red[t] = ex;
  __syncthreads();
  for (int off = 256; off > 0; off >>= 1) {
    if (t < off) red[t] += red[t + off];
    __syncthreads();
  }
  float sum = red[0];
  if (t < SS) attn[(size_t)b * SS + t] = ex / sum;
}

// ---------------- enc_context[b,h] = sum_s enc[s,b,h] * attn[b,s] ----------------
__global__ void k_ctx(const float* __restrict__ enc, const float* __restrict__ attn,
                      float* __restrict__ ctx) {
  int i = blockIdx.x * blockDim.x + threadIdx.x;
  if (i >= BB * HH) return;
  int b = i / HH, h = i - b * HH;
  float acc = 0.f;
  for (int s = 0; s < SS; ++s)
    acc += enc[(size_t)s * BB * HH + (size_t)b * HH + h] * attn[b * SS + s];
  ctx[i] = acc;
}

// ---------------- prob_ptr = sigmoid(W_ptr . concat(emb, hidden, ctx) + b_ptr) ----------------
__global__ __launch_bounds__(256) void k_ptr(const float* __restrict__ emb,
                                             const float* __restrict__ hidden,
                                             const float* __restrict__ ctx,
                                             const float* __restrict__ Wptr,
                                             const float* __restrict__ bptr,
                                             float* __restrict__ pptr) {
  __shared__ float red[256];
  int b = blockIdx.x, t = threadIdx.x;
  const int L = EE + 2 * HH + HH;   // 3584
  float acc = 0.f;
  for (int k = t; k < L; k += 256) {
    float x;
    if (k < EE) x = emb[b * EE + k];
    else if (k < EE + 2 * HH) x = hidden[(size_t)b * 2 * HH + (k - EE)];
    else x = ctx[b * HH + (k - EE - 2 * HH)];
    acc += Wptr[k] * x;
  }
  red[t] = acc;
  __syncthreads();
  for (int off = 128; off > 0; off >>= 1) {
    if (t < off) red[t] += red[t + off];
    __syncthreads();
  }
  if (t == 0) pptr[b] = sigf(red[0] + bptr[0]);
}

// ---------------- combined = concat(h, enc_context) in bf16 ----------------
__global__ void k_comb(const float* __restrict__ hidden, const float* __restrict__ ctx,
                       unsigned short* __restrict__ comb) {
  int i = blockIdx.x * blockDim.x + threadIdx.x;
  if (i >= BB * 2 * HH) return;
  int b = i / (2 * HH), k = i - b * 2 * HH;
  float v = (k < HH) ? hidden[(size_t)b * 2 * HH + k] : ctx[b * HH + (k - HH)];
  comb[i] = f2bf(v);
}

// ---------------- logits = combined @ W_out^T  (bf16 WMMA, W_out converted in-register) ----------------
__global__ __launch_bounds__(128) void k_logits(const unsigned short* __restrict__ combb,
                                                const float* __restrict__ wout,
                                                float* __restrict__ logits) {
  const int wave = threadIdx.x >> 5;
  const int lane = threadIdx.x & 31;
  const int hi = lane >> 4, n = lane & 15;
  const int vt = blockIdx.x * 4 + wave;
  if (vt >= VV / 16) return;
  const int vcol = vt * 16 + n;
  const float* wrow = wout + (size_t)vcol * (2 * HH);
  v8f acc[4];
#pragma unroll
  for (int t = 0; t < 4; ++t) {
    v8f z = {0.f, 0.f, 0.f, 0.f, 0.f, 0.f, 0.f, 0.f};
    acc[t] = z;
  }
  for (int kc = 0; kc < 2 * HH / 32; ++kc) {
    const int kb = kc * 32;
    const float4* wp = (const float4*)(wrow + kb + 8 * hi);
    const float4* wq = (const float4*)(wrow + kb + 16 + 8 * hi);
    bfrag16 bfr = frag_f(wp[0], wp[1], wq[0], wq[1]);
#pragma unroll
    for (int t = 0; t < 4; ++t) {
      const uint4* arow = (const uint4*)(combb + (size_t)(t * 16 + n) * (2 * HH));
      bfrag16 afr = frag_u(arow[(kb + 8 * hi) >> 3], arow[(kb + 16 + 8 * hi) >> 3]);
      acc[t] = __builtin_amdgcn_wmma_f32_16x16x32_bf16(
          false, afr, false, bfr, (short)0, acc[t], false, false);
    }
  }
#pragma unroll
  for (int t = 0; t < 4; ++t)
#pragma unroll
    for (int r = 0; r < 8; ++r)
      logits[(size_t)(t * 16 + 8 * hi + r) * VV + vcol] = acc[t][r];
}

// ---------------- vocab softmax stats ----------------
__global__ __launch_bounds__(512) void k_vstats(const float* __restrict__ logits,
                                                float* __restrict__ rowmax,
                                                float* __restrict__ rowsum) {
  __shared__ float red[512];
  int b = blockIdx.x, t = threadIdx.x;
  float m = -3.0e38f;
  for (int j = t; j < VV; j += 512) m = fmaxf(m, logits[(size_t)b * VV + j]);
  red[t] = m;
  __syncthreads();
  for (int off = 256; off > 0; off >>= 1) {
    if (t < off) red[t] = fmaxf(red[t], red[t + off]);
    __syncthreads();
  }
  float mx = red[0];
  __syncthreads();
  float s = 0.f;
  for (int j = t; j < VV; j += 512) s += expf(logits[(size_t)b * VV + j] - mx);
  red[t] = s;
  __syncthreads();
  for (int off = 256; off > 0; off >>= 1) {
    if (t < off) red[t] += red[t + off];
    __syncthreads();
  }
  if (t == 0) { rowmax[b] = mx; rowsum[b] = red[0]; }
}

// ---------------- out base: (1-p)*softmax(logits), zero tail ----------------
__global__ void k_base(const float* __restrict__ logits, const float* __restrict__ rowmax,
                       const float* __restrict__ rowsum, const float* __restrict__ pptr,
                       float* __restrict__ out) {
  int i = blockIdx.x * blockDim.x + threadIdx.x;
  if (i >= BB * EXTV) return;
  int b = i / EXTV, j = i - b * EXTV;
  float o = 0.f;
  if (j < VV)
    o = (1.f - pptr[b]) * expf(logits[(size_t)b * VV + j] - rowmax[b]) / rowsum[b];
  out[i] = o;
}

// ---------------- scatter-add p*attn at encoder_word_idx ----------------
__global__ void k_scatter(const int* __restrict__ widx, const float* __restrict__ attn,
                          const float* __restrict__ pptr, float* __restrict__ out) {
  int i = blockIdx.x * blockDim.x + threadIdx.x;
  if (i >= BB * SS) return;
  int b = i / SS;
  atomicAdd(&out[(size_t)b * EXTV + widx[i]], pptr[b] * attn[i]);
}

// ---------------- final log ----------------
__global__ void k_log(float* __restrict__ out) {
  int i = blockIdx.x * blockDim.x + threadIdx.x;
  if (i >= BB * EXTV) return;
  out[i] = logf(out[i] + 1e-31f);
}

extern "C" void kernel_launch(void* const* d_in, const int* in_sizes, int n_in,
                              void* d_out, int out_size, void* d_ws, size_t ws_size,
                              hipStream_t stream) {
  (void)in_sizes; (void)n_in; (void)out_size; (void)ws_size;
  const float* emb   = (const float*)d_in[0];
  const float* h0    = (const float*)d_in[1];
  const float* c0    = (const float*)d_in[2];
  const float* enc   = (const float*)d_in[3];
  const float* cov   = (const float*)d_in[4];
  const float* pctx  = (const float*)d_in[5];
  const float* Wfc   = (const float*)d_in[6];
  const float* bfc   = (const float*)d_in[7];
  const float* Wih   = (const float*)d_in[8];
  const float* Whh   = (const float*)d_in[9];
  const float* bih   = (const float*)d_in[10];
  const float* bhh   = (const float*)d_in[11];
  const float* Wq    = (const float*)d_in[12];
  const float* Wm    = (const float*)d_in[13];
  const float* battn = (const float*)d_in[14];
  const float* vattn = (const float*)d_in[15];
  const float* cw    = (const float*)d_in[16];
  const float* Wout  = (const float*)d_in[17];
  const float* Wptr  = (const float*)d_in[18];
  const float* bptr  = (const float*)d_in[19];
  const int*   mask  = (const int*)d_in[20];
  const int*   widx  = (const int*)d_in[21];

  float* out  = (float*)d_out;
  float* outH = out + (size_t)BB * EXTV;
  float* outC = outH + (size_t)BB * HH;
  float* attn = outC + (size_t)BB * HH;
  float* pptr = attn + (size_t)BB * SS;
  float* ctx  = pptr + BB;

  char* ws = (char*)d_ws;
  unsigned short* encb = (unsigned short*)ws; ws += (size_t)SS * BB * HH * 2;
  unsigned short* wmb  = (unsigned short*)ws; ws += (size_t)HH * HH * 2;
  float* dec_in = (float*)ws; ws += (size_t)BB * EE * 4;
  float* gates  = (float*)ws; ws += (size_t)BB * 4 * HH * 4;
  float* hidden = (float*)ws; ws += (size_t)BB * 2 * HH * 4;
  float* qproj  = (float*)ws; ws += (size_t)BB * HH * 4;
  float* energy = (float*)ws; ws += (size_t)BB * SS * 4;
  unsigned short* comb = (unsigned short*)ws; ws += (size_t)BB * 2 * HH * 2;
  float* logits = (float*)ws; ws += (size_t)BB * VV * 4;
  float* rowmax = (float*)ws; ws += (size_t)BB * 4;
  float* rowsum = (float*)ws; ws += (size_t)BB * 4;

  k_conv<<<(SS * BB * HH + 255) / 256, 256, 0, stream>>>(enc, encb, SS * BB * HH);
  k_conv<<<(HH * HH + 255) / 256, 256, 0, stream>>>(Wm, wmb, HH * HH);

  k_decin<<<(EE * 8 + 255) / 256, 256, 0, stream>>>(emb, pctx, Wfc, bfc, dec_in);
  k_gates<<<(4 * HH * 8 + 255) / 256, 256, 0, stream>>>(dec_in, h0, Wih, Whh, bih, bhh, gates);
  k_lstm<<<(BB * HH + 255) / 256, 256, 0, stream>>>(gates, c0, hidden, outH, outC);
  k_qproj<<<(HH * 8 + 255) / 256, 256, 0, stream>>>(hidden, Wq, qproj);

  const int esmem = BB * HH * 2 + 8 * BB * 4;   // 128KB tile + reduce buffer
  k_energy<<<SS, 256, esmem, stream>>>(encb, wmb, qproj, battn, vattn, cw, cov, mask, energy);

  k_attnsm<<<BB, 512, 0, stream>>>(energy, attn);
  k_ctx<<<(BB * HH + 255) / 256, 256, 0, stream>>>(enc, attn, ctx);
  k_ptr<<<BB, 256, 0, stream>>>(emb, hidden, ctx, Wptr, bptr, pptr);
  k_comb<<<(BB * 2 * HH + 255) / 256, 256, 0, stream>>>(hidden, ctx, comb);

  k_logits<<<(VV / 16 + 3) / 4, 128, 0, stream>>>(comb, Wout, logits);

  k_vstats<<<BB, 512, 0, stream>>>(logits, rowmax, rowsum);
  k_base<<<(BB * EXTV + 255) / 256, 256, 0, stream>>>(logits, rowmax, rowsum, pptr, out);
  k_scatter<<<(BB * SS + 255) / 256, 256, 0, stream>>>(widx, attn, pptr, out);
  k_log<<<(BB * EXTV + 255) / 256, 256, 0, stream>>>(out);
}
